// SemiFt_74749610820221
// MI455X (gfx1250) — compile-verified, matrix operand order checked by reading, not run on hardware
//
#include <hip/hip_runtime.h>
#include <hip/hip_bf16.h>

typedef __attribute__((ext_vector_type(16))) __bf16 v16bf;
typedef __attribute__((ext_vector_type(8)))  __bf16 v8bf;
typedef __attribute__((ext_vector_type(4)))  __bf16 v4bf;
typedef __attribute__((ext_vector_type(8)))  float  v8f;

#define NTOT 1029   // tokens per batch (1 cls + 4 rel + 1024 moe)
#define NB   64     // batches
#define DIM  768
#define RR   64     // low-rank dim
#define NEXP 4
#define BM   64     // tokens per workgroup tile
#define KC   32     // K chunk for the D-dim streams
#define NKC  (DIM / KC)

// LDS row pitches; all vector accesses 16B-aligned
#define PA  40      // bf16 pitch for 32-wide chunks (80B rows)
#define P64 72      // bf16 pitch for 64-wide buffers (144B rows)
#define PF  68      // fp32 pitch for full-buffer (272B rows)

#define WMMA_BF16(A, B, C) \
    __builtin_amdgcn_wmma_f32_16x16x32_bf16(false, (A), false, (B), (short)0, (C), false, false)

__device__ __forceinline__ v4bf pack4(float4 f) {
    v4bf c;
    c[0] = (__bf16)f.x; c[1] = (__bf16)f.y; c[2] = (__bf16)f.z; c[3] = (__bf16)f.w;
    return c;
}

// 16x32 bf16 operand fragment (wave32 layout per CDNA5 ISA 7.12.2):
// lane L holds row (row0 + L%16); half = L/16; VGPR v, pos p -> K = (v&3)*2+p + half*8 + (v>>2)*16
__device__ __forceinline__ v16bf ldfrag(const __bf16* __restrict__ base, int pitch,
                                        int row0, int lane) {
    const int half = lane >> 4;
    const __bf16* p = base + (size_t)(row0 + (lane & 15)) * pitch;
    v16bf f;
#pragma unroll
    for (int v = 0; v < 8; ++v) {
        const int k = ((v & 3) << 1) + (half << 3) + ((v >> 2) << 4);
        f[2 * v]     = p[k];
        f[2 * v + 1] = p[k + 1];
    }
    return f;
}

__device__ __forceinline__ float gelu_exact(float h) {
    return 0.5f * h * (1.0f + erff(h * 0.70710678118654752440f));
}

__global__ __launch_bounds__(256) void semift_fused(
    const float* __restrict__ x,   const float* __restrict__ Wd,
    const float* __restrict__ Wg,  const float* __restrict__ We,
    const float* __restrict__ be,  const float* __restrict__ Wu,
    const float* __restrict__ gmm, float* __restrict__ out)
{
    __shared__ __align__(16) __bf16 sA [2][BM * PA];  // x chunks (double buffered)
    __shared__ __align__(16) __bf16 sW [2][RR * PA];  // Wd chunks (double buffered)
    __shared__ __align__(16) __bf16 sHb[BM * P64];    // H after gelu (bf16)
    __shared__ __align__(16) __bf16 sFb[BM * P64];    // full (bf16) for up-proj
    __shared__ __align__(16) __bf16 sB64[RR * P64];   // We[e] / Wu chunk (row=out dim, col=r)
    __shared__ __align__(16) float  sFf[BM * PF];     // full (fp32)
    __shared__ float sLog[NEXP * BM];
    __shared__ int   sIdx[BM];
    __shared__ float sWgt[BM];

    const int tid  = threadIdx.x;
    const int lane = tid & 31;
    const int w    = tid >> 5;          // wave 0..7
    const int b    = blockIdx.y;
    const int tok0 = blockIdx.x * BM;

    // each wave owns tiles (mt, nt0) and (mt, nt0+1) of the 4x4 16x16 tile grid
    const int mt  = w >> 1;
    const int nt0 = (w & 1) * 2;
    // per-lane C/D element coords: N = lane%16, M rows = vgpr + 8*(lane/16)
    const int cN  = lane & 15;
    const int cM0 = (lane >> 4) * 8;

    // staging coords for 64x32 chunks: vec4 q -> row q/8, col (q%8)*4
    const int smR0 = tid >> 3;                 // j=0 row
    const int smC  = (tid & 7) * 4;            // col (float4 granule)
    int tokc0 = tok0 + smR0;       if (tokc0 >= NTOT) tokc0 = NTOT - 1;
    int tokc1 = tok0 + smR0 + 32;  if (tokc1 >= NTOT) tokc1 = NTOT - 1;
    const float* xrow0 = x + ((size_t)b * NTOT + tokc0) * DIM + smC;
    const float* xrow1 = x + ((size_t)b * NTOT + tokc1) * DIM + smC;
    const float* wrow0 = Wd + (size_t)smR0 * DIM + smC;
    const float* wrow1 = Wd + (size_t)(smR0 + 32) * DIM + smC;

    //----------------------------------------------------------------------
    // Stage 1: H = gelu(X @ Wd^T)  M=64 N=64 K=768 -- double-buffered bf16 WMMA
    //----------------------------------------------------------------------
    float4 xr0, xr1, wr0, wr1;
    // preload chunk 0
    xr0 = *(const float4*)(xrow0);  xr1 = *(const float4*)(xrow1);
    wr0 = *(const float4*)(wrow0);  wr1 = *(const float4*)(wrow1);
    *(v4bf*)&sA[0][smR0 * PA + smC]        = pack4(xr0);
    *(v4bf*)&sA[0][(smR0 + 32) * PA + smC] = pack4(xr1);
    *(v4bf*)&sW[0][smR0 * PA + smC]        = pack4(wr0);
    *(v4bf*)&sW[0][(smR0 + 32) * PA + smC] = pack4(wr1);
    __syncthreads();

    v8f acc0 = {}, acc1 = {};
    int buf = 0;
    for (int kc = 0; kc < NKC; ++kc) {
        const int k1 = (kc + 1) * KC;
        if (kc + 1 < NKC) {                     // issue next chunk's loads (b128)
            xr0 = *(const float4*)(xrow0 + k1);
            xr1 = *(const float4*)(xrow1 + k1);
            wr0 = *(const float4*)(wrow0 + k1);
            wr1 = *(const float4*)(wrow1 + k1);
            if (kc + 2 < NKC)                   // prefetch 2 chunks ahead
                __builtin_prefetch(xrow0 + k1 + KC, 0, 1);
        }
        // WMMA on current buffer
        {
            const v16bf a  = ldfrag(sA[buf], PA, mt * 16, lane);
            const v16bf b0 = ldfrag(sW[buf], PA, nt0 * 16, lane);
            const v16bf b1 = ldfrag(sW[buf], PA, (nt0 + 1) * 16, lane);
            acc0 = WMMA_BF16(a, b0, acc0);
            acc1 = WMMA_BF16(a, b1, acc1);
        }
        if (kc + 1 < NKC) {                     // convert + store into other buffer
            const int nb = buf ^ 1;
            *(v4bf*)&sA[nb][smR0 * PA + smC]        = pack4(xr0);
            *(v4bf*)&sA[nb][(smR0 + 32) * PA + smC] = pack4(xr1);
            *(v4bf*)&sW[nb][smR0 * PA + smC]        = pack4(wr0);
            *(v4bf*)&sW[nb][(smR0 + 32) * PA + smC] = pack4(wr1);
        }
        __syncthreads();
        buf ^= 1;
    }
#pragma unroll
    for (int v = 0; v < 8; ++v) {
        const int M = mt * 16 + cM0 + v;
        sHb[M * P64 + nt0 * 16 + cN]       = (__bf16)gelu_exact(acc0[v]);
        sHb[M * P64 + (nt0 + 1) * 16 + cN] = (__bf16)gelu_exact(acc1[v]);
    }
    __syncthreads();

    //----------------------------------------------------------------------
    // Stage 2: router (4 experts, top-1)  -- VALU with b128 LDS reads
    //----------------------------------------------------------------------
    {
        const int e = tid >> 6, m = tid & 63;
        const v8bf* hrow = (const v8bf*)&sHb[m * P64];
        float s = 0.0f;
#pragma unroll
        for (int r8 = 0; r8 < 8; ++r8) {
            const v8bf hv = hrow[r8];
#pragma unroll
            for (int u = 0; u < 8; ++u)
                s += (float)hv[u] * Wg[e * RR + r8 * 8 + u];
        }
        sLog[e * BM + m] = s;
    }
    __syncthreads();
    if (tid < BM) {
        const float l0 = sLog[0 * BM + tid], l1 = sLog[1 * BM + tid];
        const float l2 = sLog[2 * BM + tid], l3 = sLog[3 * BM + tid];
        int bi = 0; float bv = l0;
        if (l1 > bv) { bv = l1; bi = 1; }
        if (l2 > bv) { bv = l2; bi = 2; }
        if (l3 > bv) { bv = l3; bi = 3; }
        const float se = __expf(l0 - bv) + __expf(l1 - bv) + __expf(l2 - bv) + __expf(l3 - bv);
        sIdx[tid] = bi;
        sWgt[tid] = 1.0f / se;            // top-1 softmax prob
    }
    // init full = t (residual / cls / rel passthrough), vectorized
#pragma unroll
    for (int j = 0; j < 4; ++j) {
        const int q = j * 256 + tid;      // 1024 vec4 granules
        const int m = q >> 4, n = (q & 15) * 4;
        const v4bf hv = *(const v4bf*)&sHb[m * P64 + n];
        float4 f;
        f.x = (float)hv[0]; f.y = (float)hv[1]; f.z = (float)hv[2]; f.w = (float)hv[3];
        *(float4*)&sFf[m * PF + n] = f;
    }
    __syncthreads();

    //----------------------------------------------------------------------
    // Stage 3: experts (all 4 via WMMA; predicated top-1 writeback)
    //----------------------------------------------------------------------
    for (int e = 0; e < NEXP; ++e) {
#pragma unroll
        for (int j = 0; j < 4; ++j) {
            const int q = j * 256 + tid;
            const int row = q >> 4, col = (q & 15) * 4;
            const float4 wv = *(const float4*)&We[((size_t)e * RR + row) * RR + col];
            *(v4bf*)&sB64[row * P64 + col] = pack4(wv);
        }
        __syncthreads();
        v8f ea0 = {}, ea1 = {};
#pragma unroll
        for (int kk = 0; kk < 2; ++kk) {
            const v16bf a  = ldfrag(sHb  + kk * 32, P64, mt * 16, lane);
            const v16bf b0 = ldfrag(sB64 + kk * 32, P64, nt0 * 16, lane);
            const v16bf b1 = ldfrag(sB64 + kk * 32, P64, (nt0 + 1) * 16, lane);
            ea0 = WMMA_BF16(a, b0, ea0);
            ea1 = WMMA_BF16(a, b1, ea1);
        }
#pragma unroll
        for (int v = 0; v < 8; ++v) {
            const int M   = mt * 16 + cM0 + v;
            const int tok = tok0 + M;
            if (tok >= 5 && tok < NTOT && sIdx[M] == e) {
                const float wgt = sWgt[M];
                const int N0 = nt0 * 16 + cN, N1 = N0 + 16;
                sFf[M * PF + N0] += wgt * (ea0[v] + be[e * RR + N0]);
                sFf[M * PF + N1] += wgt * (ea1[v] + be[e * RR + N1]);
            }
        }
        __syncthreads();
    }
    // full fp32 -> bf16, vectorized
#pragma unroll
    for (int j = 0; j < 4; ++j) {
        const int q = j * 256 + tid;
        const int m = q >> 4, n = (q & 15) * 4;
        const float4 f = *(const float4*)&sFf[m * PF + n];
        *(v4bf*)&sFb[m * P64 + n] = pack4(f);
    }
    __syncthreads();

    //----------------------------------------------------------------------
    // Stage 4: OUT = FULL @ Wu^T * gamma   M=64 N=768 K=64
    //----------------------------------------------------------------------
    for (int nc = 0; nc < DIM / 64; ++nc) {
#pragma unroll
        for (int j = 0; j < 4; ++j) {
            const int q = j * 256 + tid;
            const int row = q >> 4, col = (q & 15) * 4;
            const float4 wv = *(const float4*)&Wu[(size_t)(nc * 64 + row) * RR + col];
            *(v4bf*)&sB64[row * P64 + col] = pack4(wv);
        }
        __syncthreads();
        v8f oa0 = {}, oa1 = {};
#pragma unroll
        for (int kk = 0; kk < 2; ++kk) {
            const v16bf a  = ldfrag(sFb  + kk * 32, P64, mt * 16, lane);
            const v16bf b0 = ldfrag(sB64 + kk * 32, P64, nt0 * 16, lane);
            const v16bf b1 = ldfrag(sB64 + kk * 32, P64, (nt0 + 1) * 16, lane);
            oa0 = WMMA_BF16(a, b0, oa0);
            oa1 = WMMA_BF16(a, b1, oa1);
        }
        const int d0 = nc * 64 + nt0 * 16 + cN;
        const int d1 = d0 + 16;
        const float g0 = gmm[d0], g1 = gmm[d1];
#pragma unroll
        for (int v = 0; v < 8; ++v) {
            const int M   = mt * 16 + cM0 + v;
            const int tok = tok0 + M;
            if (tok < NTOT) {
                float* o = out + ((size_t)b * NTOT + tok) * DIM;
                o[d0] = oa0[v] * g0;
                o[d1] = oa1[v] * g1;
            }
        }
        __syncthreads();
    }
}

extern "C" void kernel_launch(void* const* d_in, const int* in_sizes, int n_in,
                              void* d_out, int out_size, void* d_ws, size_t ws_size,
                              hipStream_t stream) {
    const float* x  = (const float*)d_in[0];
    const float* Wd = (const float*)d_in[1];
    const float* Wg = (const float*)d_in[2];
    const float* We = (const float*)d_in[3];
    const float* be = (const float*)d_in[4];
    const float* Wu = (const float*)d_in[5];
    const float* gm = (const float*)d_in[6];
    dim3 grid((NTOT + BM - 1) / BM, NB, 1);
    semift_fused<<<grid, 256, 0, stream>>>(x, Wd, Wg, We, be, Wu, gm, (float*)d_out);
}
